// SimpleEdgeConv_9208409883077
// MI455X (gfx1250) — matrix operand order repaired and checked by reading, not compiled, not run on hardware
//
#include <hip/hip_runtime.h>
#include <hip/hip_bf16.h>

// ---------------------------------------------------------------------------
// EdgeConv x3 for MI455X (gfx1250, wave32, WMMA).
//   max_e[(x_d - x_s)Wt + bt + x_d Wp + bp]  ==  A[d] - min_{s in in(d)} B[s]
//   A = x(Wt+Wp)+bt+bp, B = x Wt   -> node GEMMs via v_wmma_f32_16x16x32_f16
// CSR built on-device once per launch; aggregation = L2-resident min gather
// fused with LayerNorm (wave32 shfl_xor reduce) + ELU.
// GEMM: one wave computes a 16x64 strip (4 N-tiles) reusing the A fragment
// -> 2.5 b128 loads per WMMA instead of 4.
// ---------------------------------------------------------------------------

typedef __attribute__((ext_vector_type(16))) _Float16 v16h;
typedef __attribute__((ext_vector_type(8)))  _Float16 v8h;
typedef __attribute__((ext_vector_type(8)))  float    v8f;

#define GG     8
#define NN     2048
#define NODES  (GG * NN)          // 16384
#define ETOT   34816              // 32768 + 2048 self loops
#define EALL   (GG * ETOT)        // 278528
#define DHID   256
#define NOUT   512                // [A | B] concatenated
#define LN_EPS 1e-5f

// ---------------- workspace layout (bytes) ----------------
#define OFF_AB    ((size_t)0)                        // 16384*512*4  = 33554432
#define OFF_H16   (OFF_AB   + (size_t)33554432)      // 16384*256*2  = 8388608
#define OFF_WT    (OFF_H16  + (size_t)8388608)       // 512*256*2    = 262144
#define OFF_BS    (OFF_WT   + (size_t)262144)        // 256*4        = 1024
#define OFF_CNT   (OFF_BS   + (size_t)1024)          // 16384*4
#define OFF_CNT2  (OFF_CNT  + (size_t)65536)         // 16384*4
#define OFF_ROWP  (OFF_CNT2 + (size_t)65536)         // 16385*4 (padded)
#define OFF_COLX  (OFF_ROWP + (size_t)65792)         // 278528*4

// ---------------- small utility kernels ----------------
__global__ void ec_zero_i32(int* __restrict__ p, int n) {
    int i = blockIdx.x * blockDim.x + threadIdx.x;
    if (i < n) p[i] = 0;
}

__global__ void ec_cvt_f16(const float* __restrict__ x, _Float16* __restrict__ h, int n) {
    int i = blockIdx.x * blockDim.x + threadIdx.x;
    if (i < n) h[i] = (_Float16)x[i];
}

// Build WT_cat^T in f16: WT[n][k] = (n<256 ? Wt[k][n]+Wp[k][n] : Wt[k][n-256])
// and bsum[n] = bt[n] + bp[n].
__global__ void ec_prep_w(const float* __restrict__ Wt, const float* __restrict__ Wp,
                          const float* __restrict__ bt, const float* __restrict__ bp,
                          _Float16* __restrict__ WT, float* __restrict__ bsum, int K) {
    int idx = blockIdx.x * blockDim.x + threadIdx.x;
    int total = NOUT * K;
    if (idx < total) {
        int n = idx / K, k = idx - n * K;
        float w = Wt[(size_t)k * DHID + (n & (DHID - 1))];
        if (n < DHID) w += Wp[(size_t)k * DHID + n];
        WT[(size_t)n * K + k] = (_Float16)w;
    }
    if (idx < DHID) bsum[idx] = bt[idx] + bp[idx];
}

// ---------------- CSR construction (once per launch) ----------------
__global__ void ec_count(const int* __restrict__ dst, int* __restrict__ cnt) {
    int i = blockIdx.x * blockDim.x + threadIdx.x;
    if (i < EALL) {
        int g = i / ETOT;
        atomicAdd(&cnt[g * NN + dst[i]], 1);
    }
}

// Exclusive scan of 16384 counts with one 1024-thread workgroup.
__global__ __launch_bounds__(1024) void ec_scan(const int* __restrict__ cnt,
                                                int* __restrict__ rowp) {
    __shared__ int part[1024];
    int t = threadIdx.x;
    int local[16];
    int s = 0;
#pragma unroll
    for (int i = 0; i < 16; ++i) { local[i] = cnt[t * 16 + i]; s += local[i]; }
    part[t] = s;
    __syncthreads();
    for (int off = 1; off < 1024; off <<= 1) {
        int v = part[t];
        int u = (t >= off) ? part[t - off] : 0;
        __syncthreads();
        part[t] = v + u;
        __syncthreads();
    }
    int run = (t == 0) ? 0 : part[t - 1];
#pragma unroll
    for (int i = 0; i < 16; ++i) { rowp[t * 16 + i] = run; run += local[i]; }
    if (t == 1023) rowp[NODES] = run;
}

__global__ void ec_scatter(const int* __restrict__ src, const int* __restrict__ dst,
                           const int* __restrict__ rowp, int* __restrict__ fill,
                           int* __restrict__ colx) {
    int i = blockIdx.x * blockDim.x + threadIdx.x;
    if (i < EALL) {
        int g = i / ETOT;
        int d = dst[i] + g * NN;
        int s = src[i] + g * NN;
        int pos = rowp[d] + atomicAdd(&fill[d], 1);
        colx[pos] = s;
    }
}

// ---------------- GEMM: AB[16384 x 512] = H[16384 x K] * Wcat[K x 512] ----------------
// One wave per 16x64 strip (4 N-tiles share one A fragment).
// A-frag (ISA 7.12.2 16-bit A 16x32): lane l -> row (l&15), halves at
//   K = (l>>4)*8 .. +8  (VGPR0-3) and K = 16+(l>>4)*8 .. +8 (VGPR4-7).
// B-frag: lane l -> col (l&15), same K halves as A-frag (pre-transposed WT).
__global__ __launch_bounds__(256) void ec_gemm(const _Float16* __restrict__ H,
                                               const _Float16* __restrict__ WT,
                                               const float* __restrict__ bsum,
                                               float* __restrict__ AB, int K) {
    const int lane = threadIdx.x & 31;
    const int wid  = blockIdx.x * (blockDim.x >> 5) + (threadIdx.x >> 5);
    const int tg = wid & 7;    // 512/64 = 8 strip groups along N
    const int tm = wid >> 3;   // 1024 tiles along M
    const int m0 = tm << 4, n0 = tg << 6;
    const int r = lane & 15, hi = lane >> 4;

    const _Float16* __restrict__ pa  = H  + (size_t)(m0 + r) * K;
    const _Float16* __restrict__ pb0 = WT + (size_t)(n0 + r) * K;

    v8f acc[4] = {v8f{}, v8f{}, v8f{}, v8f{}};
    for (int k0 = 0; k0 < K; k0 += 32) {
        v8h a0 = *(const v8h*)(pa + k0 + hi * 8);
        v8h a1 = *(const v8h*)(pa + k0 + 16 + hi * 8);
        v16h a;
#pragma unroll
        for (int i = 0; i < 8; ++i) { a[i] = a0[i]; a[i + 8] = a1[i]; }
#pragma unroll
        for (int t = 0; t < 4; ++t) {
            const _Float16* __restrict__ pb = pb0 + (size_t)t * 16 * K;
            v8h b0 = *(const v8h*)(pb + k0 + hi * 8);
            v8h b1 = *(const v8h*)(pb + k0 + 16 + hi * 8);
            v16h b;
#pragma unroll
            for (int i = 0; i < 8; ++i) { b[i] = b0[i]; b[i + 8] = b1[i]; }
            acc[t] = __builtin_amdgcn_wmma_f32_16x16x32_f16(false, a, false, b,
                                                            (short)0, acc[t],
                                                            false, false);
        }
    }
#pragma unroll
    for (int t = 0; t < 4; ++t) {
        const int nb = n0 + t * 16;
        const float bias = (nb < DHID) ? bsum[nb + r] : 0.0f;
        float* __restrict__ out = AB + (size_t)m0 * NOUT + nb + r;
#pragma unroll
        for (int q = 0; q < 8; ++q)
            out[(size_t)(q + hi * 8) * NOUT] = acc[t][q] + bias;
    }
}

// ---------------- aggregation + LayerNorm + ELU (one wave per node) ----------------
__global__ __launch_bounds__(256) void ec_aggregate(const float* __restrict__ AB,
                                                    const int* __restrict__ rowp,
                                                    const int* __restrict__ colx,
                                                    const float* __restrict__ gamma,
                                                    const float* __restrict__ beta,
                                                    _Float16* __restrict__ Hn,
                                                    float* __restrict__ outF,
                                                    int lastLayer) {
    const int lane = threadIdx.x & 31;
    const int v = blockIdx.x * (blockDim.x >> 5) + (threadIdx.x >> 5);
    const int c0 = lane * 8;   // 32 lanes x 8 cols = 256

    float bm[8];
#pragma unroll
    for (int i = 0; i < 8; ++i) bm[i] = 3.0e38f;

    const int beg = rowp[v], end = rowp[v + 1];
    for (int e = beg; e < end; ++e) {
        const int s = colx[e];
        const float4* p = (const float4*)(AB + (size_t)s * NOUT + DHID + c0);
        float4 q0 = p[0], q1 = p[1];
        bm[0] = fminf(bm[0], q0.x); bm[1] = fminf(bm[1], q0.y);
        bm[2] = fminf(bm[2], q0.z); bm[3] = fminf(bm[3], q0.w);
        bm[4] = fminf(bm[4], q1.x); bm[5] = fminf(bm[5], q1.y);
        bm[6] = fminf(bm[6], q1.z); bm[7] = fminf(bm[7], q1.w);
    }

    const float4* pa = (const float4*)(AB + (size_t)v * NOUT + c0);
    float4 a0 = pa[0], a1 = pa[1];
    float h[8] = { a0.x - bm[0], a0.y - bm[1], a0.z - bm[2], a0.w - bm[3],
                   a1.x - bm[4], a1.y - bm[5], a1.z - bm[6], a1.w - bm[7] };

    float s = 0.f, ss = 0.f;
#pragma unroll
    for (int i = 0; i < 8; ++i) { s += h[i]; ss += h[i] * h[i]; }
#pragma unroll
    for (int m = 16; m >= 1; m >>= 1) {
        s  += __shfl_xor(s,  m, 32);
        ss += __shfl_xor(ss, m, 32);
    }
    const float mu  = s * (1.0f / 256.0f);
    const float var = ss * (1.0f / 256.0f) - mu * mu;
    const float rs  = rsqrtf(var + LN_EPS);

#pragma unroll
    for (int i = 0; i < 8; ++i) {
        float y = (h[i] - mu) * rs * gamma[c0 + i] + beta[c0 + i];
        y = (y > 0.0f) ? y : expm1f(y);
        Hn[(size_t)v * DHID + c0 + i] = (_Float16)y;
        if (lastLayer) outF[(size_t)v * DHID + c0 + i] = y;
    }
}

// ---------------- driver ----------------
extern "C" void kernel_launch(void* const* d_in, const int* in_sizes, int n_in,
                              void* d_out, int out_size, void* d_ws, size_t ws_size,
                              hipStream_t stream) {
    const float* x   = (const float*)d_in[0];
    const float* Wt[3] = { (const float*)d_in[1],  (const float*)d_in[7],  (const float*)d_in[13] };
    const float* bt[3] = { (const float*)d_in[2],  (const float*)d_in[8],  (const float*)d_in[14] };
    const float* Wp[3] = { (const float*)d_in[3],  (const float*)d_in[9],  (const float*)d_in[15] };
    const float* bp[3] = { (const float*)d_in[4],  (const float*)d_in[10], (const float*)d_in[16] };
    const float* gm[3] = { (const float*)d_in[5],  (const float*)d_in[11], (const float*)d_in[17] };
    const float* bb[3] = { (const float*)d_in[6],  (const float*)d_in[12], (const float*)d_in[18] };
    const int* src = (const int*)d_in[19];
    const int* dst = (const int*)d_in[20];
    float* out = (float*)d_out;

    char* ws = (char*)d_ws;
    float*    AB   = (float*)(ws + OFF_AB);
    _Float16* H16  = (_Float16*)(ws + OFF_H16);
    _Float16* WT16 = (_Float16*)(ws + OFF_WT);
    float*    bsum = (float*)(ws + OFF_BS);
    int*      cnt  = (int*)(ws + OFF_CNT);
    int*      fill = (int*)(ws + OFF_CNT2);
    int*      rowp = (int*)(ws + OFF_ROWP);
    int*      colx = (int*)(ws + OFF_COLX);

    // --- CSR build (deterministic result: min-reduce is order independent) ---
    ec_zero_i32<<<(NODES + 255) / 256, 256, 0, stream>>>(cnt, NODES);
    ec_zero_i32<<<(NODES + 255) / 256, 256, 0, stream>>>(fill, NODES);
    ec_count<<<(EALL + 255) / 256, 256, 0, stream>>>(dst, cnt);
    ec_scan<<<1, 1024, 0, stream>>>(cnt, rowp);
    ec_scatter<<<(EALL + 255) / 256, 256, 0, stream>>>(src, dst, rowp, fill, colx);

    // --- layer 0 input -> f16 ---
    ec_cvt_f16<<<(NODES * 128 + 255) / 256, 256, 0, stream>>>(x, H16, NODES * 128);

    const int Ks[3] = { 128, 256, 256 };
    for (int l = 0; l < 3; ++l) {
        const int K = Ks[l];
        ec_prep_w<<<(NOUT * K + 255) / 256, 256, 0, stream>>>(Wt[l], Wp[l], bt[l], bp[l],
                                                              WT16, bsum, K);
        // (16384/16)*(512/64) = 8192 waves, 8 waves/block
        ec_gemm<<<1024, 256, 0, stream>>>(H16, WT16, bsum, AB, K);
        // 16384 waves, 8 waves/block
        ec_aggregate<<<2048, 256, 0, stream>>>(AB, rowp, colx, gm[l], bb[l],
                                               H16, out, (l == 2) ? 1 : 0);
    }
}
